// Memory_66529043415393
// MI455X (gfx1250) — matrix operand order repaired and checked by reading, not compiled, not run on hardware
//
#include <hip/hip_runtime.h>
#include <stdint.h>

#define B_    16
#define CIN   128
#define NPOS  4096
#define HDIM  256
#define MDIM  1024
#define P_    (B_ * NPOS)   // 65536 positions

typedef __attribute__((ext_vector_type(16))) __bf16          v16bf;
typedef __attribute__((ext_vector_type(16))) unsigned short  v16u;
typedef __attribute__((ext_vector_type(8)))  float           v8f;

union Frag { v16u u; v16bf b; };

__device__ __forceinline__ unsigned short f2bf(float f) {
  unsigned int u = __float_as_uint(f);
  unsigned int r = u + 0x7FFFu + ((u >> 16) & 1u);   // round-to-nearest-even
  return (unsigned short)(r >> 16);
}

__device__ __forceinline__ v8f zero_v8f() {
  v8f z;
#pragma unroll
  for (int j = 0; j < 8; ++j) z[j] = 0.0f;
  return z;
}

// ---------------------------------------------------------------------------
// Async global->LDS staging (CDNA5 GLOBAL_LOAD_ASYNC_TO_LDS_B128, ASYNCcnt).
// Whole 256-thread block cooperates; 16B per lane per issue.
// ---------------------------------------------------------------------------
__device__ __forceinline__ void async_copy_block(unsigned lds_base,
                                                 const void* gsrc, int bytes) {
  unsigned long long ga = (unsigned long long)(uintptr_t)gsrc + threadIdx.x * 16;
  unsigned lo = lds_base + threadIdx.x * 16;
  for (int off = threadIdx.x * 16; off < bytes; off += 256 * 16) {
    asm volatile("global_load_async_to_lds_b128 %0, %1, off"
                 :: "v"(lo), "v"(ga) : "memory");
    lo += 256 * 16;
    ga += 256 * 16;
  }
}

__device__ __forceinline__ void wait_async_all() {
#if __has_builtin(__builtin_amdgcn_s_wait_asynccnt)
  __builtin_amdgcn_s_wait_asynccnt(0);
#else
  asm volatile("s_wait_asynccnt 0x0" ::: "memory");
#endif
}

// Wait until at most 8 async ops outstanding (double-buffer pipelining: each
// chunk is exactly 8 per-wave issues, and ASYNCcnt loads complete in order).
__device__ __forceinline__ void wait_async_le8() {
#if __has_builtin(__builtin_amdgcn_s_wait_asynccnt)
  __builtin_amdgcn_s_wait_asynccnt(8);
#else
  asm volatile("s_wait_asynccnt 0x8" ::: "memory");
#endif
}

// LDS byte offset of a __shared__ object (generic ptr low 32 bits = LDS offset).
#define LDS_OFF(p) ((unsigned)(uintptr_t)(p))

// A fragment (16x32 bf16). rowp = per-lane row base (row M = lane&15).
// ISA layout: lanes<16 hold K {0..7,16..23}; lanes>=16 hold K {8..15,24..31}.
__device__ __forceinline__ v16bf load_a_frag(const unsigned short* rowp, int k0, int lane) {
  const unsigned short* p = rowp + k0 + ((lane & 16) ? 8 : 0);
  Frag f;
#pragma unroll
  for (int j = 0; j < 16; ++j) f.u[j] = p[j + (j & 8)];
  return f.b;
}

// B fragment (32x16 bf16) from W stored as [n rows, K cols] row-major (i.e. B^T).
// Lane holds column n = lane&15; lanes<16 hold K {0..7,16..23}; lanes>=16 hold K {8..15,24..31}.
__device__ __forceinline__ v16bf load_b_frag(const unsigned short* W, int ldw,
                                             int n0, int k0, int lane) {
  const unsigned short* p = W + (size_t)(n0 + (lane & 15)) * ldw + k0 + ((lane & 16) ? 8 : 0);
  Frag f;
#pragma unroll
  for (int j = 0; j < 16; ++j) f.u[j] = p[j + (j & 8)];
  return f.b;
}

#define WMMA_BF16(A, Bm, C) \
  __builtin_amdgcn_wmma_f32_16x16x32_bf16(false, (A), false, (Bm), (short)0, (C), false, false)

// ---------------------------------------------------------------------------
// Kernel 1: per-row softmax of m1 -> mm (bf16), const[m] = sum(mm*logm),
//           plus raw m1 codebook in bf16. One block per codebook row.
// ---------------------------------------------------------------------------
__global__ void prep_softmax_kernel(const float* __restrict__ m1,
                                    unsigned short* __restrict__ mmbf,
                                    unsigned short* __restrict__ m1bf,
                                    float* __restrict__ constv) {
  __shared__ float red[8];
  int m = blockIdx.x;
  int t = threadIdx.x;                      // HDIM threads
  float v = m1[(size_t)m * HDIM + t];

  float x = v;
#pragma unroll
  for (int o = 16; o > 0; o >>= 1) x = fmaxf(x, __shfl_xor(x, o, 32));
  if ((t & 31) == 0) red[t >> 5] = x;
  __syncthreads();
  float mx = red[0];
#pragma unroll
  for (int i = 1; i < 8; ++i) mx = fmaxf(mx, red[i]);
  __syncthreads();

  float e = __expf(v - mx);
  float s = e;
#pragma unroll
  for (int o = 16; o > 0; o >>= 1) s += __shfl_xor(s, o, 32);
  if ((t & 31) == 0) red[t >> 5] = s;
  __syncthreads();
  float sum = 0.0f;
#pragma unroll
  for (int i = 0; i < 8; ++i) sum += red[i];
  __syncthreads();

  float mmv  = e / sum;
  float logm = (v - mx) - __logf(sum);
  mmbf[(size_t)m * HDIM + t] = f2bf(mmv);
  m1bf[(size_t)m * HDIM + t] = f2bf(v);

  float c = mmv * logm;
#pragma unroll
  for (int o = 16; o > 0; o >>= 1) c += __shfl_xor(c, o, 32);
  if ((t & 31) == 0) red[t >> 5] = c;
  __syncthreads();
  if (t == 0) {
    float cs = 0.0f;
#pragma unroll
    for (int i = 0; i < 8; ++i) cs += red[i];
    constv[m] = cs;
  }
}

// ---------------------------------------------------------------------------
// Kernel 2: elementwise f32 -> bf16 (for conv weights)
// ---------------------------------------------------------------------------
__global__ void cvt_bf16_kernel(const float* __restrict__ src,
                                unsigned short* __restrict__ dst, int n) {
  int i = blockIdx.x * blockDim.x + threadIdx.x;
  if (i < n) dst[i] = f2bf(src[i]);
}

// ---------------------------------------------------------------------------
// Kernel 3: transpose x [B,C,N] f32 -> xt [P=B*N, C] bf16 (LDS tiled)
// ---------------------------------------------------------------------------
__global__ void xpose_kernel(const float* __restrict__ x,
                             unsigned short* __restrict__ xt) {
  __shared__ float tile[32][33];
  int b  = blockIdx.z;
  int c0 = blockIdx.y * 32;
  int n0 = blockIdx.x * 32;
  int tx = threadIdx.x, ty = threadIdx.y;   // 32 x 8
  const float* xb = x + (size_t)b * CIN * NPOS;
#pragma unroll
  for (int i = 0; i < 4; ++i) {
    int c = c0 + ty + i * 8;
    tile[ty + i * 8][tx] = xb[(size_t)c * NPOS + n0 + tx];
  }
  __syncthreads();
  unsigned short* xtb = xt + (size_t)b * NPOS * CIN;
#pragma unroll
  for (int i = 0; i < 4; ++i) {
    int n = n0 + ty + i * 8;
    xtb[(size_t)n * CIN + c0 + tx] = f2bf(tile[tx][ty + i * 8]);
  }
}

// ---------------------------------------------------------------------------
// Kernel 4: conv1  emb[P,H] = relu(xt[P,C] * w1^T + b1)   (bf16 WMMA)
// Block = 8 waves, one 64-wide h strip (W1 strip async-staged in LDS, 16 KB),
// 128 positions per block. Wave tile: 16 pos x 64 h, K=128 in 4 steps.
// ---------------------------------------------------------------------------
__global__ void conv1_kernel(const unsigned short* __restrict__ xt,   // [P, CIN]
                             const unsigned short* __restrict__ w1,   // [H, CIN] bf16
                             const float* __restrict__ b1,
                             unsigned short* __restrict__ emb) {      // [P, H]
  __shared__ __align__(16) unsigned short sw1[64 * CIN];              // 16 KB
  int lane = threadIdx.x & 31;
  int wid  = threadIdx.x >> 5;
  int h0   = blockIdx.y * 64;
  int p0   = blockIdx.x * 128 + wid * 16;

  async_copy_block(LDS_OFF(sw1), w1 + (size_t)h0 * CIN, 64 * CIN * 2);
  wait_async_all();
  __syncthreads();

  const unsigned short* arow = xt + (size_t)(p0 + (lane & 15)) * CIN;
  v8f acc[4];
#pragma unroll
  for (int t = 0; t < 4; ++t) acc[t] = zero_v8f();

  for (int k0 = 0; k0 < CIN; k0 += 32) {
    v16bf a = load_a_frag(arow, k0, lane);
#pragma unroll
    for (int t = 0; t < 4; ++t) {
      v16bf bb = load_b_frag(sw1, CIN, t * 16, k0, lane);
      acc[t] = WMMA_BF16(a, bb, acc[t]);
    }
  }

  int prow = p0 + ((lane & 16) ? 8 : 0);
#pragma unroll
  for (int t = 0; t < 4; ++t) {
    int h = h0 + t * 16 + (lane & 15);
    float bias = b1[h];
#pragma unroll
    for (int r = 0; r < 8; ++r) {
      float val = fmaxf(acc[t][r] + bias, 0.0f);
      emb[(size_t)(prow + r) * HDIM + h] = f2bf(val);
    }
  }
}

// ---------------------------------------------------------------------------
// Kernel 5: fused scores GEMM + argmin, double-buffered async LDS pipeline.
// idx[p] = argmin_m (const[m] - emb_row[p] . mm[m])   (log-softmax cancels)
// Block = 8 waves = 128 positions. Each 64-row mm chunk = exactly 8 async
// b128 issues per wave; ASYNCcnt in-order completion => s_wait_asynccnt 8
// after issuing chunk c+1 guarantees chunk c has landed.
// ---------------------------------------------------------------------------
__global__ void scores_kernel(const unsigned short* __restrict__ emb,  // [P,H]
                              const unsigned short* __restrict__ mmbf, // [M,H]
                              const float* __restrict__ constv,        // [M]
                              int* __restrict__ idxv) {                // [P]
  __shared__ __align__(16) unsigned short smm[2][64 * HDIM];           // 2 x 32 KB
  const int NC = MDIM / 64;                                            // 16 chunks
  int lane = threadIdx.x & 31;
  int wid  = threadIdx.x >> 5;
  int p0   = blockIdx.x * 128 + wid * 16;

  const unsigned short* arow = emb + (size_t)(p0 + (lane & 15)) * HDIM;
  Frag a[8];
#pragma unroll
  for (int ks = 0; ks < 8; ++ks) a[ks].b = load_a_frag(arow, ks * 32, lane);

  float bestv[8];
  int   besti[8];
#pragma unroll
  for (int r = 0; r < 8; ++r) { bestv[r] = 3.4e38f; besti[r] = 0; }

  // prologue: start chunk 0
  async_copy_block(LDS_OFF(smm[0]), mmbf, 64 * HDIM * 2);

  for (int c = 0; c < NC; ++c) {
    if (c + 1 < NC) {
      // stream next chunk into the other buffer behind the WMMAs
      async_copy_block(LDS_OFF(smm[(c + 1) & 1]),
                       mmbf + (size_t)(c + 1) * 64 * HDIM, 64 * HDIM * 2);
      wait_async_le8();                      // chunk c complete (in-order)
    } else {
      wait_async_all();
    }
    __syncthreads();                         // publish chunk c block-wide

    const unsigned short* cur = smm[c & 1];
#pragma unroll
    for (int mt = 0; mt < 4; ++mt) {
      v8f acc = zero_v8f();
#pragma unroll
      for (int ks = 0; ks < 8; ++ks) {
        v16bf bb = load_b_frag(cur, HDIM, mt * 16, ks * 32, lane);
        acc = WMMA_BF16(a[ks].b, bb, acc);
      }
      int m = c * 64 + mt * 16 + (lane & 15);
      float cst = constv[m];
#pragma unroll
      for (int r = 0; r < 8; ++r) {
        float kl = cst - acc[r];
        if (kl < bestv[r]) { bestv[r] = kl; besti[r] = m; }
      }
    }
    __syncthreads();                         // all waves done reading buf[c&1]
  }

  // argmin across the 16 lanes of each half-wave (m lives in lane&15).
#pragma unroll
  for (int r = 0; r < 8; ++r) {
    float bv = bestv[r];
    int   bi = besti[r];
#pragma unroll
    for (int off = 1; off < 16; off <<= 1) {
      float ov = __shfl_xor(bv, off, 32);
      int   oi = __shfl_xor(bi, off, 32);
      if (ov < bv || (ov == bv && oi < bi)) { bv = ov; bi = oi; }   // first-min tie-break
    }
    bestv[r] = bv; besti[r] = bi;
  }
  if ((lane & 15) == 0) {
    int pr = p0 + ((lane & 16) ? 8 : 0);
#pragma unroll
    for (int r = 0; r < 8; ++r) idxv[pr + r] = besti[r];
  }
}

// ---------------------------------------------------------------------------
// Kernel 6: conv2  out[b,h,n] = [emb ; m1[idx]] . w2^T + b2   (bf16 WMMA)
// Block = 8 waves, one 64-wide h_out strip (W2 strip async-staged, 64 KB LDS).
// K = 512 (first 256 from emb, last 256 from gathered codebook rows).
// ---------------------------------------------------------------------------
__global__ void conv2_kernel(const unsigned short* __restrict__ emb,   // [P,H]
                             const unsigned short* __restrict__ m1bf,  // [M,H]
                             const int* __restrict__ idxv,             // [P]
                             const unsigned short* __restrict__ w2,    // [H, 2H] bf16
                             const float* __restrict__ b2,
                             float* __restrict__ out) {                // [B,H,N]
  __shared__ __align__(16) unsigned short sw2[64 * 2 * HDIM];          // 64 KB
  int lane = threadIdx.x & 31;
  int wid  = threadIdx.x >> 5;
  int h0   = blockIdx.y * 64;
  int p0   = blockIdx.x * 128 + wid * 16;

  async_copy_block(LDS_OFF(sw2), w2 + (size_t)h0 * 2 * HDIM, 64 * 2 * HDIM * 2);

  int pr = p0 + (lane & 15);
  const unsigned short* embrow = emb  + (size_t)pr * HDIM;
  const unsigned short* selrow = m1bf + (size_t)idxv[pr] * HDIM;   // per-lane gather

  wait_async_all();
  __syncthreads();

  v8f acc[4];
#pragma unroll
  for (int t = 0; t < 4; ++t) acc[t] = zero_v8f();

  for (int k0 = 0; k0 < HDIM; k0 += 32) {
    v16bf ae = load_a_frag(embrow, k0, lane);
    v16bf as = load_a_frag(selrow, k0, lane);
#pragma unroll
    for (int t = 0; t < 4; ++t) {
      v16bf be = load_b_frag(sw2, 2 * HDIM, t * 16, k0, lane);
      acc[t] = WMMA_BF16(ae, be, acc[t]);
      v16bf bs = load_b_frag(sw2, 2 * HDIM, t * 16, HDIM + k0, lane);
      acc[t] = WMMA_BF16(as, bs, acc[t]);
    }
  }

  int b  = p0 / NPOS;
  int n0 = p0 - b * NPOS;
#pragma unroll
  for (int t = 0; t < 4; ++t) {
    int h = h0 + t * 16 + (lane & 15);
    float bias = b2[h];
    float* op = out + ((size_t)b * HDIM + h) * NPOS + n0 + ((lane & 16) ? 8 : 0);
#pragma unroll
    for (int r = 0; r < 8; ++r) op[r] = acc[t][r] + bias;   // 8 contiguous floats/lane
  }
}

// ---------------------------------------------------------------------------
extern "C" void kernel_launch(void* const* d_in, const int* in_sizes, int n_in,
                              void* d_out, int out_size, void* d_ws, size_t ws_size,
                              hipStream_t stream) {
  const float* x   = (const float*)d_in[0];
  const float* m1  = (const float*)d_in[1];
  const float* w1f = (const float*)d_in[2];
  const float* b1  = (const float*)d_in[3];
  const float* w2f = (const float*)d_in[4];
  const float* b2  = (const float*)d_in[5];
  float* out = (float*)d_out;

  char* ws = (char*)d_ws;
  size_t o = 0;
  auto alloc = [&](size_t bytes) {
    char* p = ws + o;
    o = (o + bytes + 255) & ~(size_t)255;
    return p;
  };
  unsigned short* xt    = (unsigned short*)alloc((size_t)P_ * CIN * 2);        // 16 MB
  unsigned short* emb   = (unsigned short*)alloc((size_t)P_ * HDIM * 2);       // 32 MB
  unsigned short* mmbf  = (unsigned short*)alloc((size_t)MDIM * HDIM * 2);     // 512 KB
  unsigned short* m1bf  = (unsigned short*)alloc((size_t)MDIM * HDIM * 2);     // 512 KB
  unsigned short* w1    = (unsigned short*)alloc((size_t)HDIM * CIN * 2);      // 64 KB
  unsigned short* w2    = (unsigned short*)alloc((size_t)HDIM * 2 * HDIM * 2); // 256 KB
  float*          cst   = (float*)alloc((size_t)MDIM * 4);
  int*            idxv  = (int*)alloc((size_t)P_ * 4);

  prep_softmax_kernel<<<MDIM, HDIM, 0, stream>>>(m1, mmbf, m1bf, cst);
  cvt_bf16_kernel<<<(HDIM * CIN + 255) / 256, 256, 0, stream>>>(w1f, w1, HDIM * CIN);
  cvt_bf16_kernel<<<(HDIM * 2 * HDIM + 255) / 256, 256, 0, stream>>>(w2f, w2, HDIM * 2 * HDIM);
  xpose_kernel<<<dim3(NPOS / 32, CIN / 32, B_), dim3(32, 8), 0, stream>>>(x, xt);

  conv1_kernel<<<dim3(P_ / 128, HDIM / 64), 256, 0, stream>>>(xt, w1, b1, emb);
  scores_kernel<<<P_ / 128, 256, 0, stream>>>(emb, mmbf, cst, idxv);
  conv2_kernel<<<dim3(P_ / 128, HDIM / 64), 256, 0, stream>>>(emb, m1bf, idxv, w2, b2, out);
}
